// NVFP4LinearKernel_70841190580411
// MI455X (gfx1250) — 8-bit matrix operand order repaired and checked by reading, compiled, not run on hardware
//
#include <hip/hip_runtime.h>
#include <hip/hip_fp16.h>

typedef __attribute__((ext_vector_type(16))) int   v16i;
typedef __attribute__((ext_vector_type(8)))  float v8f;

#define BM 128
#define BN 256
#define BK 128
#define LDST 144
#define NBUF 3
#define A_BYTES (BM * LDST)
#define B_BYTES (BN * LDST)
#define SMEM_BYTES (NBUF * (A_BYTES + B_BYTES))

#if defined(__has_builtin)
#  if __has_builtin(__builtin_amdgcn_cvt_pk_fp8_f32)
#    define HAVE_HW_FP8_CVT 1
#  endif
#endif
#ifndef HAVE_HW_FP8_CVT
#  define HAVE_HW_FP8_CVT 0
#endif

__device__ __forceinline__ unsigned int f32_to_e4m3(float x) {
    unsigned int sign = (__float_as_uint(x) >> 24) & 0x80u;
    float a = fminf(fabsf(x), 448.0f);
    unsigned int au = __float_as_uint(a);
    int e = (int)((au >> 23) & 0xFFu) - 127;
    if (e < -6) e = -6;
    float q = rintf(a * __int_as_float((unsigned int)(130 - e) << 23));
    q = q * __int_as_float((unsigned int)(124 + e) << 23);
    unsigned int qu = __float_as_uint(q);
    int re = (int)((qu >> 23) & 0xFFu) - 127;
    unsigned int mag;
    if (re < -6) mag = (unsigned int)(q * 512.0f);
    else         mag = ((unsigned int)(re + 7) << 3) | ((qu >> 20) & 0x7u);
    return mag | sign;
}

__device__ __forceinline__ unsigned int pack2_e4m3(float f0, float f1) {
#if HAVE_HW_FP8_CVT
    f0 = fminf(fmaxf(f0, -448.0f), 448.0f);
    f1 = fminf(fmaxf(f1, -448.0f), 448.0f);
    return (unsigned int)__builtin_amdgcn_cvt_pk_fp8_f32(f0, f1, 0, false) & 0xFFFFu;
#else
    return f32_to_e4m3(f0) | (f32_to_e4m3(f1) << 8);
#endif
}

__global__ void fp8lin_reset_kernel(unsigned int* absmax) {
    if (threadIdx.x == 0) *absmax = 0u;
}

__global__ void fp8lin_quant_x_kernel(const uint4* __restrict__ x,
                                      uint2* __restrict__ xq,
                                      unsigned int* __restrict__ absmax,
                                      long n8) {
    long stride = (long)gridDim.x * blockDim.x;
    float mx = 0.0f;
    for (long i = (long)blockIdx.x * blockDim.x + threadIdx.x; i < n8; i += stride) {
        uint4 u = x[i];
        unsigned int w[4] = {u.x, u.y, u.z, u.w};
        float2 f[4];
#pragma unroll
        for (int j = 0; j < 4; ++j) {
            __half2 h2 = *reinterpret_cast<const __half2*>(&w[j]);
            f[j] = __half22float2(h2);
            mx = fmaxf(mx, fmaxf(fabsf(f[j].x), fabsf(f[j].y)));
        }
        unsigned int lo = pack2_e4m3(f[0].x, f[0].y) | (pack2_e4m3(f[1].x, f[1].y) << 16);
        unsigned int hi = pack2_e4m3(f[2].x, f[2].y) | (pack2_e4m3(f[3].x, f[3].y) << 16);
        xq[i] = make_uint2(lo, hi);
    }
#pragma unroll
    for (int off = 16; off > 0; off >>= 1)
        mx = fmaxf(mx, __shfl_xor(mx, off, 32));
    if ((threadIdx.x & 31) == 0)
        atomicMax(absmax, __float_as_uint(mx));
}

__global__ void fp8lin_quant_w_kernel(const float4* __restrict__ w,
                                      unsigned int* __restrict__ wq,
                                      long n4) {
    long stride = (long)gridDim.x * blockDim.x;
    for (long i = (long)blockIdx.x * blockDim.x + threadIdx.x; i < n4; i += stride) {
        float4 f = w[i];
        wq[i] = pack2_e4m3(f.x, f.y) | (pack2_e4m3(f.z, f.w) << 16);
    }
}

__device__ __forceinline__ void fp8lin_stage_async(
        const unsigned char* __restrict__ Aq,
        const unsigned char* __restrict__ Bq,
        unsigned int ldsA, unsigned int ldsB,
        int blockM, int blockN, int K, int k0, int tid) {
#pragma unroll
    for (int i = 0; i < 4; ++i) {
        int idx = tid + i * 256;
        int row = idx >> 3, chunk = idx & 7;
        unsigned int l = ldsA + (unsigned int)(row * LDST + chunk * 16);
        unsigned int g = (unsigned int)((blockM + row) * K + k0 + chunk * 16);
        asm volatile("global_load_async_to_lds_b128 %0, %1, %2"
                     :: "v"(l), "v"(g), "s"(Aq) : "memory");
    }
#pragma unroll
    for (int i = 0; i < 8; ++i) {
        int idx = tid + i * 256;
        int row = idx >> 3, chunk = idx & 7;
        unsigned int l = ldsB + (unsigned int)(row * LDST + chunk * 16);
        unsigned int g = (unsigned int)((blockN + row) * K + k0 + chunk * 16);
        asm volatile("global_load_async_to_lds_b128 %0, %1, %2"
                     :: "v"(l), "v"(g), "s"(Bq) : "memory");
    }
}

__global__ __launch_bounds__(256)
void fp8lin_gemm_kernel(const unsigned char* __restrict__ Aq,
                        const unsigned char* __restrict__ Bq,
                        const unsigned int* __restrict__ absmax,
                        const float* __restrict__ wscale,
                        const __half* __restrict__ bias,
                        __half* __restrict__ out,
                        int M, int N, int K) {
    extern __shared__ unsigned char smem[];

    const int tid   = threadIdx.x;
    const int wave  = tid >> 5;
    const int lane  = tid & 31;
    const int waveM = wave >> 2;
    const int waveN = wave & 3;
    const int l16   = lane & 15;
    const int half  = lane >> 4;

    const int blockM = blockIdx.y * BM;
    const int blockN = blockIdx.x * BN;

    const unsigned int smemBase = (unsigned int)(uintptr_t)smem;
    const unsigned int ldsA0 = smemBase;
    const unsigned int ldsB0 = smemBase + NBUF * A_BYTES;

    v8f acc[4][4] = {};

    const int nIter = K / BK;

    fp8lin_stage_async(Aq, Bq, ldsA0, ldsB0, blockM, blockN, K, 0, tid);
    if (nIter > 1) {
        fp8lin_stage_async(Aq, Bq, ldsA0 + A_BYTES, ldsB0 + B_BYTES,
                           blockM, blockN, K, BK, tid);
        asm volatile("s_wait_asynccnt 0xc" ::: "memory");
    } else {
        asm volatile("s_wait_asynccnt 0x0" ::: "memory");
    }
    __syncthreads();

    int cur = 0;
    int issueBuf = 2;
    for (int it = 0; it < nIter; ++it) {
        const bool moreIssue = (it + 2) < nIter;
        if (moreIssue)
            fp8lin_stage_async(Aq, Bq,
                               ldsA0 + (unsigned int)(issueBuf * A_BYTES),
                               ldsB0 + (unsigned int)(issueBuf * B_BYTES),
                               blockM, blockN, K, (it + 2) * BK, tid);

        const unsigned char* Acur = smem + cur * A_BYTES;
        const unsigned char* Bcur = smem + NBUF * A_BYTES + cur * B_BYTES;

        v16i afrag[4];
#pragma unroll
        for (int mi = 0; mi < 4; ++mi) {
            const unsigned char* abase =
                Acur + (waveM * 64 + mi * 16 + l16) * LDST + half * 16;
#pragma unroll
            for (int c = 0; c < 8; ++c) {
                uint2 d = *(const uint2*)(abase + (c >> 1) * 32 + (c & 1) * 8);
                afrag[mi][2 * c]     = (int)d.x;
                afrag[mi][2 * c + 1] = (int)d.y;
            }
        }

#pragma unroll
        for (int ni = 0; ni < 4; ++ni) {
            const unsigned char* bbase =
                Bcur + (waveN * 64 + ni * 16 + l16) * LDST + half * 16;
            v16i bfrag;
#pragma unroll
            for (int q = 0; q < 4; ++q) {
                uint4 d = *(const uint4*)(bbase + q * 32);
                bfrag[4 * q]     = (int)d.x;
                bfrag[4 * q + 1] = (int)d.y;
                bfrag[4 * q + 2] = (int)d.z;
                bfrag[4 * q + 3] = (int)d.w;
            }
#pragma unroll
            for (int mi = 0; mi < 4; ++mi)
                acc[mi][ni] = __builtin_amdgcn_wmma_f32_16x16x128_fp8_fp8(
                    afrag[mi], bfrag, (short)0, acc[mi][ni], false, false);
        }

        if (moreIssue) asm volatile("s_wait_asynccnt 0xc" ::: "memory");
        else           asm volatile("s_wait_asynccnt 0x0" ::: "memory");
        __syncthreads();

        cur      = (cur == NBUF - 1) ? 0 : cur + 1;
        issueBuf = (issueBuf == NBUF - 1) ? 0 : issueBuf + 1;
    }

    float am = fmaxf(__uint_as_float(*absmax), 1e-12f);
    float s  = am * (1.0f / 448.0f) * wscale[0];

#pragma unroll
    for (int ni = 0; ni < 4; ++ni) {
        int n = blockN + waveN * 64 + ni * 16 + l16;
        float bv = __half2float(bias[n]);
#pragma unroll
        for (int mi = 0; mi < 4; ++mi) {
            int mBase = blockM + waveM * 64 + mi * 16 + half * 8;
#pragma unroll
            for (int r = 0; r < 8; ++r) {
                float v = acc[mi][ni][r] * s + bv;
                out[(size_t)(mBase + r) * N + n] = __float2half(v);
            }
        }
    }
}

extern "C" void kernel_launch(void* const* d_in, const int* in_sizes, int n_in,
                              void* d_out, int out_size, void* d_ws, size_t ws_size,
                              hipStream_t stream) {
    const __half* x      = (const __half*)d_in[0];
    const float*  w      = (const float*)d_in[1];
    const float*  wscale = (const float*)d_in[2];
    const __half* bias   = (const __half*)d_in[3];
    __half*       out    = (__half*)d_out;

    const long xN = in_sizes[0];
    const long wN = in_sizes[1];
    const int  N  = in_sizes[3];
    const long K  = wN / N;
    const long M  = xN / K;

    unsigned char* ws     = (unsigned char*)d_ws;
    unsigned int*  absmax = (unsigned int*)ws;
    unsigned char* xq     = ws + 256;
    unsigned char* wq     = ws + 256 + (size_t)M * K;

    fp8lin_reset_kernel<<<1, 32, 0, stream>>>(absmax);

    const long n8 = (M * K) / 8;
    fp8lin_quant_x_kernel<<<1024, 256, 0, stream>>>(
        (const uint4*)x, (uint2*)xq, absmax, n8);

    const long n4 = ((long)N * K) / 4;
    fp8lin_quant_w_kernel<<<1024, 256, 0, stream>>>(
        (const float4*)w, (unsigned int*)wq, n4);

    dim3 grid((N + BN - 1) / BN, (M + BM - 1) / BM);
    fp8lin_gemm_kernel<<<grid, 256, SMEM_BYTES, stream>>>(
        xq, wq, absmax, wscale, bias, out, (int)M, (int)N, (int)K);
}
